// MultiHeadAttentionWithROPE_88029649699027
// MI455X (gfx1250) — matrix operand order repaired and checked by reading, not compiled, not run on hardware
//
#include <hip/hip_runtime.h>
#include <hip/hip_bf16.h>

// ---------------------------------------------------------------------------
// MultiHeadAttention + RoPE for MI455X (gfx1250, wave32, WMMA)
//   B=64, S=200, D=1024, H=16, hd=64
// Kernel 0: f32 -> f16 conversion of x and W (one-time, keeps GEMM loop lean)
// Kernel 1: QKV projection GEMM (f16 WMMA, f32 accum) + bias + RoPE epilogue,
//           staging f16 tiles global->LDS via global_load_async_to_lds_b128.
// Kernel 2: fused attention per (b, h, 16-query tile): QK^T WMMA, masked
//           softmax in LDS, P@V WMMA, f32 output.
// ---------------------------------------------------------------------------

#define B_    64
#define S_    200
#define D_    1024
#define H_    16
#define HD    64
#define SPAD  208   // padded seq for Q/K rows   (13 * 16)
#define SPADV 224   // padded seq for V^T cols   ( 7 * 32)

typedef _Float16 v16h __attribute__((ext_vector_type(16)));
typedef _Float16 v8h  __attribute__((ext_vector_type(8)));
typedef _Float16 v4h  __attribute__((ext_vector_type(4)));
typedef float    v8f  __attribute__((ext_vector_type(8)));

#define NEG_INF (-__builtin_inff())

// ---- CDNA5 async global->LDS copy (16B per lane), tracked by ASYNCcnt -----
__device__ __forceinline__ void async_copy_16B(const void* g, const void* lds) {
  unsigned l = (unsigned)(unsigned long long)lds;  // low 32b of shared aperture = LDS offset
  asm volatile("global_load_async_to_lds_b128 %0, %1, off"
               :: "v"(l), "v"(g) : "memory");
}
__device__ __forceinline__ void wait_async0() {
  asm volatile("s_wait_asynccnt 0" ::: "memory");
}

__device__ __forceinline__ v16h cat8(v8h lo, v8h hi) {
  return __builtin_shufflevector(lo, hi, 0,1,2,3,4,5,6,7,8,9,10,11,12,13,14,15);
}

// ---------------------------------------------------------------------------
// Kernel 0: elementwise f32 -> f16 (vectorized x4)
// ---------------------------------------------------------------------------
__global__ __launch_bounds__(256)
void cvt_f16_kernel(const float* __restrict__ src, _Float16* __restrict__ dst,
                    int n4)
{
  int i = blockIdx.x * 256 + threadIdx.x;
  if (i < n4) {
    float4 v = ((const float4*)src)[i];
    v4h o;
    o[0] = (_Float16)v.x; o[1] = (_Float16)v.y;
    o[2] = (_Float16)v.z; o[3] = (_Float16)v.w;
    *(v4h*)(dst + (size_t)i * 4) = o;
  }
}

// ---------------------------------------------------------------------------
// Kernel 1: qkv = x @ W^T + b  ->  RoPE(q), RoPE(k), v^T  (all f16)
//   grid (200, 48), block 256.  Block tile: M=64, N=64, BK=64 (2 wmma chunks)
// ---------------------------------------------------------------------------
__global__ __launch_bounds__(256)
void qkv_rope_kernel(const _Float16* __restrict__ xh, const _Float16* __restrict__ Wh,
                     const float* __restrict__ bias,
                     _Float16* __restrict__ Qh, _Float16* __restrict__ Kh,
                     _Float16* __restrict__ VT)
{
  __shared__ _Float16 Ash[64][64];   // x tile   [m][k]  (8 KB)
  __shared__ _Float16 Bsh[64][64];   // W tile   [n][k]  (8 KB; W row == B column)
  __shared__ float    Cs[64][65];    // f32 result tile

  const int tid  = threadIdx.x;
  const int lane = tid & 31;
  const int w    = tid >> 5;     // 8 waves
  const int wm   = w & 3;        // wave M tile (4 x 16 rows)
  const int wn   = w >> 2;       // wave N tile (2 x 32 cols)
  const int bm   = blockIdx.x * 64;
  const int bn   = blockIdx.y * 64;

  const int mrow = lane & 15;
  const int half = lane >> 4;
  const int kh   = half * 16;

  v8f acc[2] = { {0.f,0.f,0.f,0.f,0.f,0.f,0.f,0.f},
                 {0.f,0.f,0.f,0.f,0.f,0.f,0.f,0.f} };

  for (int k0 = 0; k0 < D_; k0 += 64) {
    // ---- stage A (64x64 f16) and B (64x64 f16) via async DMA to LDS ----
#pragma unroll
    for (int t = 0; t < 2; ++t) {
      int f = tid + t * 256;                 // 0..511 16B slots (8 per row)
      int r = f >> 3, cc = (f & 7) << 3;     // cc in halves
      async_copy_16B(xh + (size_t)(bm + r) * D_ + k0 + cc, &Ash[r][cc]);
    }
#pragma unroll
    for (int t = 0; t < 2; ++t) {
      int f = tid + t * 256;
      int r = f >> 3, cc = (f & 7) << 3;
      async_copy_16B(Wh + (size_t)(bn + r) * D_ + k0 + cc, &Bsh[r][cc]);
    }
    wait_async0();
    __syncthreads();

#pragma unroll
    for (int c = 0; c < 2; ++c) {
      // A fragment: row m, K = {kb..kb+15} within 32-chunk
      const _Float16* ar = &Ash[wm * 16 + mrow][c * 32];
      v16h a = cat8(*(const v8h*)(ar + kh), *(const v8h*)(ar + kh + 8));
#pragma unroll
      for (int nt = 0; nt < 2; ++nt) {
        const _Float16* br = &Bsh[wn * 32 + nt * 16 + mrow][c * 32 + half * 16];
        v16h bf = *(const v16h*)br;          // contiguous 32B per lane
        acc[nt] = __builtin_amdgcn_wmma_f32_16x16x32_f16(
            false, a, false, bf, (short)0, acc[nt], false, false);
      }
    }
    __syncthreads();
  }

  // ---- stage C tile to LDS (C layout: m = v + 8*half, n = lane&15) ----
#pragma unroll
  for (int nt = 0; nt < 2; ++nt)
#pragma unroll
    for (int v = 0; v < 8; ++v)
      Cs[wm * 16 + v + half * 8][wn * 32 + nt * 16 + mrow] = acc[nt][v];
  __syncthreads();

  // ---- epilogue: bias + RoPE + scatter (64 rows x 32 column-pairs) ----
  const float LN1E4_64 = 0.14391156f;   // ln(10000)/64
#pragma unroll
  for (int t = 0; t < 8; ++t) {
    int idx = tid + t * 256;            // 0..2047
    int r   = idx >> 5;                 // row in tile
    int pj  = idx & 31;                 // pair index
    int row = bm + r;
    int bi  = row / S_;
    int s   = row - bi * S_;
    int c0  = bn + pj * 2;
    float x0 = Cs[r][pj * 2]     + bias[c0];
    float x1 = Cs[r][pj * 2 + 1] + bias[c0 + 1];
    int hh   = c0 / 192;
    int tt   = c0 - hh * 192;
    int kind = tt >> 6;                 // 0=q 1=k 2=v
    int d    = tt & 63;                 // even
    if (kind == 2) {
      size_t base = ((size_t)(bi * H_ + hh) * HD + d) * SPADV + s;
      VT[base]         = (_Float16)x0;
      VT[base + SPADV] = (_Float16)x1;
    } else {
      float invf = __expf(-(float)d * LN1E4_64);  // 10000^(-d/64)
      float sn, cs;
      __sincosf((float)s * invf, &sn, &cs);
      float r0 = x0 * cs - x1 * sn;
      float r1 = x0 * sn + x1 * cs;
      _Float16* dst = (kind == 0 ? Qh : Kh) +
                      ((size_t)(bi * H_ + hh) * SPAD + s) * HD + d;
      dst[0] = (_Float16)r0;
      dst[1] = (_Float16)r1;
    }
  }
}

// ---------------------------------------------------------------------------
// Kernel 2: attention for one (b, h, 16-query tile) per wave.
//   grid (13, 1024), block 32 (one wave32).
// ---------------------------------------------------------------------------
__global__ __launch_bounds__(32)
void attn_kernel(const _Float16* __restrict__ Qh, const _Float16* __restrict__ Kh,
                 const _Float16* __restrict__ VT, float* __restrict__ out)
{
  __shared__ float    Ls[16][SPAD];    // logits (then exp)
  __shared__ _Float16 Ps[16][SPADV];   // probabilities, f16

  const int lane = threadIdx.x;
  const int qt   = blockIdx.x;         // 0..12
  const int bh   = blockIdx.y;         // b*16 + h
  const int b    = bh >> 4;
  const int h    = bh & 15;
  const int qb   = qt * 16;

  const int mrow = lane & 15;
  const int half = lane >> 4;

  // ---- Q fragments, held in registers for all key tiles (K=64 -> 2 chunks)
  const _Float16* qrow = Qh + ((size_t)bh * SPAD + (qb + mrow)) * HD;
  v16h aq[2];
#pragma unroll
  for (int c = 0; c < 2; ++c) {
    v8h lo = *(const v8h*)(qrow + c * 32 + half * 16);
    v8h hi = *(const v8h*)(qrow + c * 32 + half * 16 + 8);
    aq[c] = cat8(lo, hi);
  }

  // ---- logits = (Q K^T) / 8, causal-masked, to LDS ----
  const float scale = 0.125f;          // 1/sqrt(64)
  for (int kt = 0; kt < 13; ++kt) {
    const int nb = kt * 16;
    v8f acc = {0.f,0.f,0.f,0.f,0.f,0.f,0.f,0.f};
#pragma unroll
    for (int c = 0; c < 2; ++c) {
      const _Float16* kp =
          Kh + ((size_t)bh * SPAD + (nb + mrow)) * HD + c * 32 + half * 16;
      v16h bk = *(const v16h*)kp;      // contiguous 32B per lane
      acc = __builtin_amdgcn_wmma_f32_16x16x32_f16(
          false, aq[c], false, bk, (short)0, acc, false, false);
    }
    const int kpos = nb + mrow;        // key position for this lane's column
#pragma unroll
    for (int v = 0; v < 8; ++v) {
      int qp = qb + v + half * 8;
      Ls[v + half * 8][kpos] =
          (kpos <= qp && kpos < S_) ? acc[v] * scale : NEG_INF;
    }
  }
  __syncthreads();

  // ---- softmax: 2 lanes per row (halves merged with shfl_xor 16) ----
  const int cbase = half * 104;
  float mx = NEG_INF;
  for (int j = 0; j < 104; ++j) mx = fmaxf(mx, Ls[mrow][cbase + j]);
  mx = fmaxf(mx, __shfl_xor(mx, 16, 32));
  float sum = 0.f;
  for (int j = 0; j < 104; ++j) {
    float e = __expf(Ls[mrow][cbase + j] - mx);
    Ls[mrow][cbase + j] = e;
    sum += e;
  }
  sum += __shfl_xor(sum, 16, 32);
  float inv = 1.0f / sum;              // diagonal element guarantees sum >= 1
  for (int j = 0; j < 104; ++j)
    Ps[mrow][cbase + j] = (_Float16)(Ls[mrow][cbase + j] * inv);
#pragma unroll
  for (int j = 0; j < 8; ++j)          // zero-pad K to 224
    Ps[mrow][SPAD + half * 8 + j] = (_Float16)0.f;
  __syncthreads();

  // ---- out[16x64] = P[16x224] @ V[224x64] ----
  v8f accO[4] = { {0.f,0.f,0.f,0.f,0.f,0.f,0.f,0.f},
                  {0.f,0.f,0.f,0.f,0.f,0.f,0.f,0.f},
                  {0.f,0.f,0.f,0.f,0.f,0.f,0.f,0.f},
                  {0.f,0.f,0.f,0.f,0.f,0.f,0.f,0.f} };
  for (int c = 0; c < 7; ++c) {
    v8h lo = *(const v8h*)&Ps[mrow][c * 32 + half * 16];
    v8h hi = *(const v8h*)&Ps[mrow][c * 32 + half * 16 + 8];
    v16h ap = cat8(lo, hi);
#pragma unroll
    for (int nt = 0; nt < 4; ++nt) {
      const _Float16* vp =
          VT + ((size_t)bh * HD + nt * 16 + mrow) * SPADV + c * 32 + half * 16;
      v16h bv = *(const v16h*)vp;      // contiguous 32B per lane (V is transposed)
      accO[nt] = __builtin_amdgcn_wmma_f32_16x16x32_f16(
          false, ap, false, bv, (short)0, accO[nt], false, false);
    }
  }
#pragma unroll
  for (int nt = 0; nt < 4; ++nt)
#pragma unroll
    for (int v = 0; v < 8; ++v) {
      int qp = qb + v + half * 8;
      if (qp < S_)
        out[((size_t)b * S_ + qp) * D_ + h * HD + nt * 16 + mrow] = accO[nt][v];
    }
}

// ---------------------------------------------------------------------------
extern "C" void kernel_launch(void* const* d_in, const int* in_sizes, int n_in,
                              void* d_out, int out_size, void* d_ws, size_t ws_size,
                              hipStream_t stream) {
  const float* x    = (const float*)d_in[0];
  // d_in[1] = mask (exact causal tril; recomputed analytically in-kernel)
  const float* W    = (const float*)d_in[2];
  const float* bias = (const float*)d_in[3];
  float* out        = (float*)d_out;

  const size_t QK_ELEMS = (size_t)B_ * H_ * SPAD * HD;   // 13,631,488
  const size_t VT_ELEMS = (size_t)B_ * H_ * HD * SPADV;  // 14,680,064
  const size_t X_ELEMS  = (size_t)B_ * S_ * D_;          // 13,107,200
  const size_t W_ELEMS  = (size_t)3 * D_ * D_;           //  3,145,728

  _Float16* Qh = (_Float16*)d_ws;
  _Float16* Kh = Qh + QK_ELEMS;
  _Float16* VT = Kh + QK_ELEMS;
  _Float16* xh = VT + VT_ELEMS;
  _Float16* Wh = xh + X_ELEMS;

  // zero Q/K/V padded regions (graph-capture-safe memset node)
  hipMemsetAsync(d_ws, 0, (2 * QK_ELEMS + VT_ELEMS) * sizeof(_Float16), stream);

  cvt_f16_kernel<<<(int)(X_ELEMS / 4 / 256), 256, 0, stream>>>(x, xh, (int)(X_ELEMS / 4));
  cvt_f16_kernel<<<(int)(W_ELEMS / 4 / 256), 256, 0, stream>>>(W, Wh, (int)(W_ELEMS / 4));

  qkv_rope_kernel<<<dim3((B_ * S_) / 64, (3 * D_) / 64), 256, 0, stream>>>(
      xh, Wh, bias, Qh, Kh, VT);
  attn_kernel<<<dim3(SPAD / 16, B_ * H_), 32, 0, stream>>>(Qh, Kh, VT, out);
}
